// GFAPF_29317446762626
// MI455X (gfx1250) — matrix operand order repaired and checked by reading, not compiled, not run on hardware
//
#include <hip/hip_runtime.h>

// ---- CDNA5 (gfx1250) bf16 WMMA, wave32 ----
typedef __attribute__((ext_vector_type(16))) __bf16 v16bf;
typedef __attribute__((ext_vector_type(8)))  float  v8f;
typedef unsigned int u32x4 __attribute__((ext_vector_type(4)));
typedef int          i32x8 __attribute__((ext_vector_type(8)));
typedef int          i32x4 __attribute__((ext_vector_type(4)));

#define D_MODEL 512
#define D_K     64
#define SEQ     512
#define KLEN    127   // 2*64 - 1

__device__ __forceinline__ v8f wmma_bf16(v16bf a, v16bf b, v8f c) {
  // D = A(16x32) * B(32x16) + C, f32 accumulate
  return __builtin_amdgcn_wmma_f32_16x16x32_bf16(
      false, a, false, b, (short)0, c, false, false);
}

// ---------------------------------------------------------------------------
// TDM: load a (rows x 64) bf16 tile, row-major stride 64, into LDS with one
// pad DWORD after every 32 DWORDs (i.e. LDS row stride = 66 bf16 elements).
// D# group0: count=1 | lds_addr | global_addr | type=2 ("image").
// D# group1: data_size=1(2B), pad_enable, pad_interval=4(32dw), pad_amount=0(1dw),
//            tensor_dim0=64, tensor_dim1=rows, tile_dim0=64, tile_dim1=rows,
//            tensor_dim0_stride=64.
// Toolchain here is the 6-arg clang-23 form:
//   (u32x4 g0, i32x8 g1, i32x4 g2, i32x4 g3, i32x8 extra, i32 cpol)
// ---------------------------------------------------------------------------
__device__ __forceinline__ void tdm_load_tile_bf16(unsigned lds_off,
                                                   const __bf16* gptr,
                                                   int rows) {
  unsigned long long ga = (unsigned long long)(uintptr_t)gptr;
  u32x4 g0;
  g0[0] = 1u;                                        // count = 1
  g0[1] = lds_off;                                   // LDS byte address
  g0[2] = (unsigned)(ga & 0xffffffffu);              // global_addr[31:0]
  g0[3] = (unsigned)((ga >> 32) & 0x01ffffffu) | 0x80000000u; // ga[56:32] | type=2
  i32x8 g1;
  g1[0] = (1 << 16) | (1 << 20) | (4 << 22);         // data_size=2B, pad_en, interval=32dw, amount=1dw
  g1[1] = (64 << 16);                                // tensor_dim0[15:0] = 64
  g1[2] = (rows << 16);                              // tensor_dim0[31:16]=0 | tensor_dim1[15:0]=rows
  g1[3] = (64 << 16);                                // tensor_dim1[31:16]=0 | tile_dim0=64
  g1[4] = rows;                                      // tile_dim1=rows | tile_dim2=0
  g1[5] = 64;                                        // tensor_dim0_stride[31:0] = 64
  g1[6] = 0;                                         // stride hi | tensor_dim1_stride lo
  g1[7] = 0;
  i32x4 z4 = {0, 0, 0, 0};
  i32x8 z8 = {0, 0, 0, 0, 0, 0, 0, 0};
  __builtin_amdgcn_tensor_load_to_lds(g0, g1, z4, z4, z8, 0);
}

__device__ __forceinline__ unsigned lds_offset_of(const void* p) {
  // generic LDS address low 32 bits == LDS byte offset (aperture in high bits)
  return (unsigned)(uintptr_t)p;
}

// Build a 32x16 bf16 B-fragment from one LDS row at element offset `base`
// (uniform parity across the wave): 9 aligned dword reads + alignbit for odd.
__device__ __forceinline__ v16bf load_bfrag(const __bf16* rowptr, int base) {
  const unsigned* rp = (const unsigned*)rowptr;
  const int half = base >> 1;
  unsigned w[9];
  #pragma unroll
  for (int i = 0; i < 9; ++i) w[i] = rp[half + i];
  union { v16bf v; unsigned u[8]; } r;
  if (base & 1) {
    #pragma unroll
    for (int i = 0; i < 8; ++i) r.u[i] = (w[i] >> 16) | (w[i + 1] << 16);
  } else {
    #pragma unroll
    for (int i = 0; i < 8; ++i) r.u[i] = w[i];
  }
  return r.v;
}

// ---------------------------------------------------------------------------
// Kernel 1: fused QKV projection. (B*S,512) x (512,64) x3, bf16 WMMA.
// Grid 64 x 128 thr (4 waves); wave w owns n-units nt = 3w..3w+2.
// ---------------------------------------------------------------------------
__global__ __launch_bounds__(128) void qkv_kernel(
    const float* __restrict__ x,
    const float* __restrict__ Wq, const float* __restrict__ bq,
    const float* __restrict__ Wk, const float* __restrict__ bk,
    const float* __restrict__ Wv, const float* __restrict__ bv,
    __bf16* __restrict__ qo, __bf16* __restrict__ ko, __bf16* __restrict__ vo)
{
  __shared__ __bf16 xs[16][D_MODEL + 2];
  const int tid  = threadIdx.x;
  const int lane = tid & 31, wave = tid >> 5;
  const int hi   = lane >> 4, ln = lane & 15;
  const int row0 = blockIdx.x * 16;

  for (int idx = tid; idx < 16 * D_MODEL; idx += 128) {
    int r = idx >> 9, c = idx & (D_MODEL - 1);
    xs[r][c] = (__bf16)x[(size_t)(row0 + r) * D_MODEL + c];
  }
  __syncthreads();

  v8f acc[3] = {};
  for (int kt = 0; kt < D_MODEL / 32; ++kt) {
    const int k0 = kt * 32;
    v16bf a;
    #pragma unroll
    for (int e = 0; e < 16; ++e) {
      int ka = e + 16 * hi;                       // A 16x32 bf16 layout
      a[e] = xs[ln][k0 + ka];
    }
    #pragma unroll
    for (int c = 0; c < 3; ++c) {
      const int nt = wave * 3 + c;
      const int mat = nt >> 2, n0 = (nt & 3) * 16;
      const float* W = (mat == 0) ? Wq : ((mat == 1) ? Wk : Wv);
      v16bf bm;
      #pragma unroll
      for (int e = 0; e < 16; ++e) {
        int kb = e + 16 * hi;                      // B 32x16 bf16 layout
        bm[e] = (__bf16)W[(size_t)(k0 + kb) * D_K + n0 + ln];
      }
      acc[c] = wmma_bf16(a, bm, acc[c]);
    }
  }
  #pragma unroll
  for (int c = 0; c < 3; ++c) {
    const int nt = wave * 3 + c;
    const int mat = nt >> 2, n0 = (nt & 3) * 16;
    const float* bb = (mat == 0) ? bq : ((mat == 1) ? bk : bv);
    __bf16* dst = (mat == 0) ? qo : ((mat == 1) ? ko : vo);
    const float bias = bb[n0 + ln];
    #pragma unroll
    for (int e = 0; e < 8; ++e)
      dst[(size_t)(row0 + e + 8 * hi) * D_K + n0 + ln] = (__bf16)(acc[c][e] + bias);
  }
}

// ---------------------------------------------------------------------------
// Kernel 2: fused conv-score + online logsumexp + s@v (flash style).
// Grid = B*(S/16)*4 = 256 blocks, 128 thr (4 waves).
// ---------------------------------------------------------------------------
__global__ __launch_bounds__(128) void score_kernel(
    const __bf16* __restrict__ qb, const __bf16* __restrict__ kb,
    const __bf16* __restrict__ vb, float* __restrict__ p)
{
  __shared__ __bf16 qs[16][D_K + 2];     // Q tile (TDM staged)
  __shared__ __bf16 kp[4][16][194];      // per-wave reversed zero-padded K rows
  __shared__ __bf16 vs[4][32][D_K + 2];  // per-wave V rows (TDM staged)
  __shared__ __bf16 ss[4][16][34];       // per-wave s tile (16 x 32)

  const int tid  = threadIdx.x;
  const int lane = tid & 31, wave = tid >> 5;
  const int hi   = lane >> 4, ln = lane & 15;
  const int blk  = blockIdx.x;
  const int jr   = blk & 3;
  const int it   = (blk >> 2) & 31;
  const int b    = blk >> 7;
  const int i0   = b * SEQ + it * 16;

  // Stage Q tile via Tensor Data Mover (one wave issues; barrier publishes).
  if (wave == 0) {
    tdm_load_tile_bf16(lds_offset_of(&qs[0][0]), qb + (size_t)i0 * D_K, 16);
  }
  // Zero kp borders once (cols [0,64) and [128,194) stay zero forever).
  for (int idx = tid; idx < 4 * 16 * 194; idx += 128) {
    int wv = idx / (16 * 194);
    int rem = idx - wv * (16 * 194);
    int r = rem / 194, c = rem - r * 194;
    if (c < 64 || c >= 128) kp[wv][r][c] = (__bf16)0.0f;
  }
  if (wave == 0) __builtin_amdgcn_s_wait_tensorcnt(0);
  __syncthreads();

  // Fixed A fragments: Q tile columns a = 0..31 and 32..63
  v16bf aq0, aq1;
  #pragma unroll
  for (int e = 0; e < 16; ++e) {
    int ka = e + 16 * hi;
    aq0[e] = qs[ln][ka];
    aq1[e] = qs[ln][32 + ka];
  }

  v8f pacc[4] = {};

  #pragma unroll 1
  for (int h = 0; h < 2; ++h) {
    const int jt = b * SEQ + jr * 128 + wave * 16 + h * 64;  // K/V tile base row
    // V rows via TDM into this wave's vs half (row stride 66 via pad).
    tdm_load_tile_bf16(lds_offset_of(&vs[wave][h * 16][0]),
                       vb + (size_t)jt * D_K, 16);
    // kp[r][64+c] = k[jt+r][63-c]  =>  B[a][n] = kp[n][(127-t)+a] = k[n][t-a]
    for (int idx = lane; idx < 16 * 64; idx += 32) {
      int r = idx >> 6, c = idx & 63;
      kp[wave][r][64 + c] = kb[(size_t)(jt + r) * D_K + (63 - c)];
    }
    __builtin_amdgcn_s_wait_tensorcnt(0);
    __syncthreads();

    v8f mv, se, sz;
    #pragma unroll
    for (int e = 0; e < 8; ++e) { mv[e] = -__builtin_inff(); se[e] = 0.f; sz[e] = 0.f; }

    // software-pipelined: WMMA for t overlaps softmax update for t-1
    v8f dprev = {};
    float scalprev = 0.f;
    for (int t = 0; t < KLEN; ++t) {
      const float nd   = (t < 64) ? (float)(t + 1) : (float)(127 - t);
      const float scal = (float)(t + 2) / nd;
      v8f d = {};
      const __bf16* krow = &kp[wave][ln][0];
      if (t <= 94) {                        // fragment a in [0,31] has support
        v16bf bm = load_bfrag(krow, (127 - t) + 16 * hi);
        d = wmma_bf16(aq0, bm, d);
      }
      if (t >= 32) {                        // fragment a in [32,63] has support
        v16bf bm = load_bfrag(krow, (127 - t + 32) + 16 * hi);
        d = wmma_bf16(aq1, bm, d);
      }
      if (t > 0) {
        #pragma unroll
        for (int e = 0; e < 8; ++e) {       // online logsumexp + sum over t
          float z  = dprev[e] * scalprev;
          float nm = fmaxf(mv[e], z);
          se[e] = se[e] * __expf(mv[e] - nm) + __expf(z - nm);
          mv[e] = nm;
          sz[e] += z;
        }
      }
      dprev = d; scalprev = scal;
    }
    #pragma unroll
    for (int e = 0; e < 8; ++e) {           // drain pipeline (t = KLEN-1)
      float z  = dprev[e] * scalprev;
      float nm = fmaxf(mv[e], z);
      se[e] = se[e] * __expf(mv[e] - nm) + __expf(z - nm);
      mv[e] = nm;
      sz[e] += z;
    }
    #pragma unroll
    for (int e = 0; e < 8; ++e) {
      float s = sz[e] - 127.0f * (mv[e] + __logf(se[e]));
      ss[wave][e + 8 * hi][h * 16 + ln] = (__bf16)s;   // transpose via LDS
    }
    __syncthreads();
  }

  // p += s(16x32) @ v(32x64): one WMMA per 16-col n-tile
  {
    v16bf as;
    #pragma unroll
    for (int e = 0; e < 16; ++e) {
      int ka = e + 16 * hi;
      as[e] = ss[wave][ln][ka];
    }
    #pragma unroll
    for (int nt = 0; nt < 4; ++nt) {
      v16bf bm;
      #pragma unroll
      for (int e = 0; e < 16; ++e) bm[e] = vs[wave][e + 16 * hi][nt * 16 + ln];
      pacc[nt] = wmma_bf16(as, bm, pacc[nt]);
    }
  }
  #pragma unroll
  for (int nt = 0; nt < 4; ++nt) {
    #pragma unroll
    for (int e = 0; e < 8; ++e)
      atomicAdd(&p[(size_t)(i0 + e + 8 * hi) * D_K + nt * 16 + ln], pacc[nt][e]);
  }
}

// ---------------------------------------------------------------------------
// Kernel 3: out = p @ Wp + bp. (1024,64)x(64,64). Grid 64 x 128 thr.
// ---------------------------------------------------------------------------
__global__ __launch_bounds__(128) void proj_kernel(
    const float* __restrict__ pg, const float* __restrict__ Wp,
    const float* __restrict__ bp, float* __restrict__ out)
{
  __shared__ __bf16 ps[16][D_K + 2];
  const int tid  = threadIdx.x;
  const int lane = tid & 31, wave = tid >> 5;
  const int hi   = lane >> 4, ln = lane & 15;
  const int row0 = blockIdx.x * 16;

  for (int idx = tid; idx < 16 * D_K; idx += 128) {
    int r = idx >> 6, c = idx & 63;
    ps[r][c] = (__bf16)pg[(size_t)(row0 + r) * D_K + c];
  }
  __syncthreads();

  v16bf a0, a1;
  #pragma unroll
  for (int e = 0; e < 16; ++e) {
    int ka = e + 16 * hi;
    a0[e] = ps[ln][ka];
    a1[e] = ps[ln][32 + ka];
  }
  const int n0 = wave * 16;
  v16bf b0, b1;
  #pragma unroll
  for (int e = 0; e < 16; ++e) {
    int kb = e + 16 * hi;
    b0[e] = (__bf16)Wp[(size_t)kb * D_K + n0 + ln];
    b1[e] = (__bf16)Wp[(size_t)(32 + kb) * D_K + n0 + ln];
  }
  v8f d = {};
  d = wmma_bf16(a0, b0, d);
  d = wmma_bf16(a1, b1, d);
  const float bias = bp[n0 + ln];
  #pragma unroll
  for (int e = 0; e < 8; ++e)
    out[(size_t)(row0 + e + 8 * hi) * D_K + n0 + ln] = d[e] + bias;
}

// ---------------------------------------------------------------------------
extern "C" void kernel_launch(void* const* d_in, const int* in_sizes, int n_in,
                              void* d_out, int out_size, void* d_ws, size_t ws_size,
                              hipStream_t stream) {
  (void)in_sizes; (void)n_in; (void)out_size; (void)ws_size;
  const float* x  = (const float*)d_in[0];
  const float* Wq = (const float*)d_in[1];
  const float* bq = (const float*)d_in[2];
  const float* Wk = (const float*)d_in[3];
  const float* bk = (const float*)d_in[4];
  const float* Wv = (const float*)d_in[5];
  const float* bv = (const float*)d_in[6];
  const float* Wp = (const float*)d_in[7];
  const float* bp = (const float*)d_in[8];
  float* out = (float*)d_out;

  char* ws = (char*)d_ws;
  __bf16* qb = (__bf16*)(ws);                 // 1024*64*2 = 128 KB
  __bf16* kb = (__bf16*)(ws + 131072);
  __bf16* vb = (__bf16*)(ws + 262144);
  float*  p  = (float*) (ws + 393216);        // 1024*64*4 = 256 KB

  (void)hipMemsetAsync(p, 0, (size_t)1024 * 64 * sizeof(float), stream);
  qkv_kernel<<<64, 128, 0, stream>>>(x, Wq, bq, Wk, bk, Wv, bv, qb, kb, vb);
  score_kernel<<<256, 128, 0, stream>>>(qb, kb, vb, p);
  proj_kernel<<<64, 128, 0, stream>>>(p, Wp, bp, out);
}
